// MultiHeadSelfAttention_60215441490172
// MI455X (gfx1250) — matrix operand order repaired and checked by reading, not compiled, not run on hardware
//
#include <hip/hip_runtime.h>
#include <stdint.h>
#include <stddef.h>

#define S_LEN  2048
#define DMODEL 1024
#define NHEAD  16
#define HDIM   64
#define BATCH  2

#define BM 128
#define BN 128
#define BK 32

typedef __attribute__((ext_vector_type(16))) __bf16 v16bf;
typedef __attribute__((ext_vector_type(8)))  float  v8f;
typedef __attribute__((ext_vector_type(4)))  unsigned int v4u;
typedef __attribute__((ext_vector_type(8)))  int v8i;
typedef __attribute__((ext_vector_type(4)))  int v4i;

#if defined(__AMDGCN__) && __has_builtin(__builtin_amdgcn_tensor_load_to_lds)
#define HAVE_TDM 1
#else
#define HAVE_TDM 0
#endif

union FragB16 { uint4 q[2]; v16bf v; };

__device__ __forceinline__ uint16_t f2bf(float f) {
  union { float f; uint32_t u; } w; w.f = f;
  uint32_t r = w.u + 0x7FFFu + ((w.u >> 16) & 1u);   // round-to-nearest-even
  return (uint16_t)(r >> 16);
}

#if HAVE_TDM
// Issue a TDM 2D tile load (bf16 elements): tile (tile0 x tile1) from a tensor
// with row length tdim0 (elements), row stride stride0 (elements) -> LDS.
// 6-arg builtin form (clang-23 / therock-10.0 headers).
__device__ __forceinline__
void tdm_load_2d_bf16(uint32_t lds_addr, const void* gptr,
                      uint32_t tdim0, uint32_t tdim1,
                      uint32_t tile0, uint32_t tile1, uint64_t stride0) {
  const uint64_t ga = (uint64_t)gptr;
  v4u g0;
  g0.x = 1u;                                              // count=1 (valid D#)
  g0.y = lds_addr;                                        // bits 63:32
  g0.z = (uint32_t)ga;                                    // bits 95:64
  g0.w = (uint32_t)((ga >> 32) & 0x01FFFFFFu) | (2u << 30); // ga[56:32] | type=2
  v8i g1;
  g1[0] = (int)(1u << 16);                                // data_size=1 (2 bytes)
  g1[1] = (int)((tdim0 & 0xFFFFu) << 16);                 // tensor_dim0[15:0]
  g1[2] = (int)(((tdim0 >> 16) & 0xFFFFu) | ((tdim1 & 0xFFFFu) << 16));
  g1[3] = (int)(((tdim1 >> 16) & 0xFFFFu) | ((tile0 & 0xFFFFu) << 16));
  g1[4] = (int)(tile1 & 0xFFFFu);                         // tile_dim1; tile_dim2=0
  g1[5] = (int)(uint32_t)(stride0 & 0xFFFFFFFFu);         // dim0_stride[31:0]
  g1[6] = (int)(uint32_t)((stride0 >> 32) & 0xFFFFu);     // dim0_stride[47:32]
  g1[7] = 0;
  const v4i z4 = {0, 0, 0, 0};
  const v8i z8 = {0, 0, 0, 0, 0, 0, 0, 0};
  __builtin_amdgcn_tensor_load_to_lds(g0, g1, z4, z4, z8, 0);
}
#endif

// ---------------------------------------------------------------------------
// Kernel 1: fp32 -> bf16 conversion, 4 elements/thread
// ---------------------------------------------------------------------------
__global__ __launch_bounds__(256)
void f32_to_bf16_kernel(const float* __restrict__ src, uint16_t* __restrict__ dst, int n4) {
  int i = blockIdx.x * blockDim.x + threadIdx.x;
  if (i >= n4) return;
  float4 f = ((const float4*)src)[i];
  union { uint16_t u[4]; uint2 v; } o;
  o.u[0] = f2bf(f.x); o.u[1] = f2bf(f.y); o.u[2] = f2bf(f.z); o.u[3] = f2bf(f.w);
  ((uint2*)dst)[i] = o.v;
}

// ---------------------------------------------------------------------------
// Shared 128x128-tile bf16 GEMM mainloop: C = A[128xK] * Bt[128xK]^T
// 256 threads = 8 waves (4x2); each wave computes 32x64 via 2x4 WMMA tiles.
// Global->LDS staging uses CDNA5 async-to-LDS (ASYNCcnt) with double buffer.
// ---------------------------------------------------------------------------
__device__ __forceinline__
void gemm128x128(const uint16_t* __restrict__ A, const uint16_t* __restrict__ Bt,
                 int K, int rowA0, int rowB0, v8f acc[2][4]) {
  __shared__ __align__(16) uint16_t As[2][BM * BK];
  __shared__ __align__(16) uint16_t Bs[2][BM * BK];
  const int tid   = threadIdx.x;
  const int lane  = tid & 31;
  const int wave  = tid >> 5;
  const int wm    = (wave >> 1) * 32;      // 0,32,64,96
  const int wn    = (wave & 1) * 64;       // 0,64
  const int lrow  = tid >> 1;              // 0..127 tile row fetched by this thread
  const int lhalf = (tid & 1) * 16;        // 0 or 16 (K element offset)
  const int m     = lane & 15;
  const int half  = lane >> 4;

  const uint32_t asb = (uint32_t)(size_t)&As[0][0];   // LDS byte offsets
  const uint32_t bsb = (uint32_t)(size_t)&Bs[0][0];
  const uint32_t lofs = (uint32_t)(lrow * BK + lhalf) * 2;
  const uint64_t gA0 = (uint64_t)(A  + (size_t)(rowA0 + lrow) * K + lhalf);
  const uint64_t gB0 = (uint64_t)(Bt + (size_t)(rowB0 + lrow) * K + lhalf);

  // ---- prologue: async-load K-tile 0 into buffer 0 ----
  {
    const uint32_t la = asb + lofs, lb = bsb + lofs;
    asm volatile("global_load_async_to_lds_b128 %0, %1, off"           :: "v"(la), "v"(gA0) : "memory");
    asm volatile("global_load_async_to_lds_b128 %0, %1, off offset:16" :: "v"(la), "v"(gA0) : "memory");
    asm volatile("global_load_async_to_lds_b128 %0, %1, off"           :: "v"(lb), "v"(gB0) : "memory");
    asm volatile("global_load_async_to_lds_b128 %0, %1, off offset:16" :: "v"(lb), "v"(gB0) : "memory");
  }
  asm volatile("s_wait_asynccnt 0x0" ::: "memory");
  __syncthreads();

  int buf = 0;
  for (int k0 = 0; k0 < K; k0 += BK) {
    // ---- issue async loads for next K-tile into the other buffer ----
    if (k0 + BK < K) {
      const uint32_t bo = (uint32_t)(buf ^ 1) * (BM * BK * 2);
      const uint32_t la = asb + bo + lofs, lb = bsb + bo + lofs;
      const uint64_t ga = gA0 + (uint64_t)(k0 + BK) * 2;
      const uint64_t gb = gB0 + (uint64_t)(k0 + BK) * 2;
      asm volatile("global_load_async_to_lds_b128 %0, %1, off"           :: "v"(la), "v"(ga) : "memory");
      asm volatile("global_load_async_to_lds_b128 %0, %1, off offset:16" :: "v"(la), "v"(ga) : "memory");
      asm volatile("global_load_async_to_lds_b128 %0, %1, off"           :: "v"(lb), "v"(gb) : "memory");
      asm volatile("global_load_async_to_lds_b128 %0, %1, off offset:16" :: "v"(lb), "v"(gb) : "memory");
    }
    // ---- fragments from current buffer + 8 WMMAs ----
    const uint16_t* Ab = &As[buf][0];
    const uint16_t* Bb = &Bs[buf][0];
    FragB16 fa[2], fb[4];
    #pragma unroll
    for (int t = 0; t < 2; ++t) {
      const uint16_t* pa = &Ab[(wm + t * 16 + m) * BK + half * 8];
      fa[t].q[0] = *(const uint4*)pa;
      fa[t].q[1] = *(const uint4*)(pa + 16);
    }
    #pragma unroll
    for (int t = 0; t < 4; ++t) {
      const uint16_t* pb = &Bb[(wn + t * 16 + m) * BK + half * 8];
      fb[t].q[0] = *(const uint4*)pb;
      fb[t].q[1] = *(const uint4*)(pb + 16);
    }
    #pragma unroll
    for (int tm = 0; tm < 2; ++tm)
      #pragma unroll
      for (int tn = 0; tn < 4; ++tn)
        acc[tm][tn] = __builtin_amdgcn_wmma_f32_16x16x32_bf16(
            false, fa[tm].v, false, fb[tn].v, (short)0, acc[tm][tn], false, false);

    asm volatile("s_wait_asynccnt 0x0" ::: "memory");   // next tile landed in LDS
    __syncthreads();
    buf ^= 1;
  }
}

// ---------------------------------------------------------------------------
// Kernel 2: QKV projection.  qkv[m,e] = x[m,:] . Wqkv[e,:]
// Scatters q,k -> (B,H,S,hd), v -> transposed (B,H,hd,S).
// ---------------------------------------------------------------------------
__global__ __launch_bounds__(256)
void qkv_gemm_kernel(const uint16_t* __restrict__ xb, const uint16_t* __restrict__ wqkvb,
                     uint16_t* __restrict__ qb, uint16_t* __restrict__ kb,
                     uint16_t* __restrict__ vtb) {
  v8f acc[2][4] = {};
  const int rowA0 = blockIdx.x * BM;
  const int rowB0 = blockIdx.y * BN;
  gemm128x128(xb, wqkvb, DMODEL, rowA0, rowB0, acc);

  const int lane  = threadIdx.x & 31;
  const int wave  = threadIdx.x >> 5;
  const int wm    = (wave >> 1) * 32;
  const int wn    = (wave & 1) * 64;
  const int col   = lane & 15;
  const int rbase = (lane >> 4) * 8;
  #pragma unroll
  for (int tm = 0; tm < 2; ++tm)
    #pragma unroll
    for (int tn = 0; tn < 4; ++tn)
      #pragma unroll
      for (int r = 0; r < 8; ++r) {
        const int gm = rowA0 + wm + tm * 16 + rbase + r;   // 0..4095
        const int gn = rowB0 + wn + tn * 16 + col;         // 0..3071
        const uint16_t val = f2bf(acc[tm][tn][r]);
        const int b = gm >> 11, s = gm & (S_LEN - 1);
        if (gn < DMODEL) {
          const int h = gn >> 6, d = gn & 63;
          qb[(((size_t)(b * NHEAD + h) * S_LEN) + s) * HDIM + d] = val;
        } else if (gn < 2 * DMODEL) {
          const int e = gn - DMODEL, h = e >> 6, d = e & 63;
          kb[(((size_t)(b * NHEAD + h) * S_LEN) + s) * HDIM + d] = val;
        } else {
          const int e = gn - 2 * DMODEL, h = e >> 6, d = e & 63;
          vtb[(((size_t)(b * NHEAD + h) * HDIM) + d) * S_LEN + s] = val;
        }
      }
}

// ---------------------------------------------------------------------------
// Kernel 3: flash attention. Block = 8 waves x 16 query rows for one (b,h).
// K/V tiles staged once per block into LDS: K via async-to-LDS (contiguous),
// V via Tensor Data Mover 2D descriptor (strided), double-buffered.
// ---------------------------------------------------------------------------
__global__ __launch_bounds__(256)
void attention_kernel(const uint16_t* __restrict__ qb, const uint16_t* __restrict__ kb,
                      const uint16_t* __restrict__ vtb, const unsigned char* __restrict__ mask,
                      uint16_t* __restrict__ aob) {
  __shared__ __align__(16) uint16_t Ks[2][64 * 64];    // [j_local][d]
  __shared__ __align__(16) uint16_t Vs[2][64 * 64];    // [d][j_local]
  __shared__ __align__(16) uint16_t Pst[8][16 * 64];   // wave-private P tiles
  const float scale = 0.125f;                          // 1/sqrt(64)
  const int tid  = threadIdx.x;
  const int wave = tid >> 5;
  const int lane = tid & 31;
  const int bh   = blockIdx.y;                         // 0..31
  const int b    = bh >> 4, h = bh & 15;
  const int q0   = (blockIdx.x * 8 + wave) * 16;
  const int m    = lane & 15;
  const int half = lane >> 4;
  const int rbase = half * 8;

  const uint16_t* Qp = qb  + ((size_t)bh * S_LEN + q0) * HDIM;
  const uint16_t* Kb = kb  + (size_t)bh * S_LEN * HDIM;
  const uint16_t* Vb = vtb + (size_t)bh * HDIM * S_LEN;
  const unsigned char* maskp = mask + (size_t)b * S_LEN;
  uint16_t* Pw = Pst[wave];

  const uint32_t ksBase = (uint32_t)(size_t)&Ks[0][0];
  const uint32_t vsBase = (uint32_t)(size_t)&Vs[0][0];

  // Stage key/value tile starting at j0 into LDS buffer bf.
  auto stage = [&](int j0, int bf) {
    {  // K tile: 64x64 bf16, contiguous 8KB -> per-thread async b128 pairs
      const uint64_t ga = (uint64_t)(Kb + (size_t)j0 * HDIM + tid * 16);
      const uint32_t la = ksBase + (uint32_t)bf * (64 * 64 * 2) + (uint32_t)tid * 32;
      asm volatile("global_load_async_to_lds_b128 %0, %1, off"           :: "v"(la), "v"(ga) : "memory");
      asm volatile("global_load_async_to_lds_b128 %0, %1, off offset:16" :: "v"(la), "v"(ga) : "memory");
    }
#if HAVE_TDM
    if (tid < 32)   // one wave issues the strided 2D V tile via the TDM
      tdm_load_2d_bf16(vsBase + (uint32_t)bf * (64 * 64 * 2), Vb + j0,
                       S_LEN, HDIM, 64, 64, S_LEN);
#else
    {  // fallback: async b128 pairs, 4 threads per Vt row
      const int d = tid >> 2, qo = (tid & 3) * 16;
      const uint64_t ga = (uint64_t)(Vb + (size_t)d * S_LEN + j0 + qo);
      const uint32_t la = vsBase + (uint32_t)bf * (64 * 64 * 2) + (uint32_t)(d * 64 + qo) * 2;
      asm volatile("global_load_async_to_lds_b128 %0, %1, off"           :: "v"(la), "v"(ga) : "memory");
      asm volatile("global_load_async_to_lds_b128 %0, %1, off offset:16" :: "v"(la), "v"(ga) : "memory");
    }
#endif
  };

  // Q fragments stay resident in registers for the whole kernel.
  FragB16 aq[2];
  #pragma unroll
  for (int ks = 0; ks < 2; ++ks) {
    const uint16_t* p = Qp + m * HDIM + ks * 32 + half * 8;
    aq[ks].q[0] = *(const uint4*)p;
    aq[ks].q[1] = *(const uint4*)(p + 16);
  }

  v8f O[4] = {};
  float rmax[8], rsum[8];
  #pragma unroll
  for (int r = 0; r < 8; ++r) { rmax[r] = -3.4028235e38f; rsum[r] = 0.f; }

  stage(0, 0);
  asm volatile("s_wait_asynccnt 0x0" ::: "memory");
#if HAVE_TDM
  __builtin_amdgcn_s_wait_tensorcnt(0);
#endif
  __syncthreads();

  int bf = 0;
  for (int j0 = 0; j0 < S_LEN; j0 += 64) {
    if (j0 + 64 < S_LEN) stage(j0 + 64, bf ^ 1);     // prefetch next tile

    // ---- S = Q . K^T (4 N-subtiles x 2 K-steps), K-frags from LDS ----
    float sv[4][8];
    #pragma unroll
    for (int sub = 0; sub < 4; ++sub) {
      v8f sacc = {};
      #pragma unroll
      for (int ks = 0; ks < 2; ++ks) {
        FragB16 fk;
        const uint16_t* p = &Ks[bf][(sub * 16 + m) * 64 + ks * 32 + half * 8];
        fk.q[0] = *(const uint4*)p;
        fk.q[1] = *(const uint4*)(p + 16);
        sacc = __builtin_amdgcn_wmma_f32_16x16x32_bf16(
            false, aq[ks].v, false, fk.v, (short)0, sacc, false, false);
      }
      const bool mk = maskp[j0 + sub * 16 + m] != 0;
      #pragma unroll
      for (int r = 0; r < 8; ++r)
        sv[sub][r] = mk ? -3.4028235e38f : sacc[r] * scale;
    }
    // ---- online softmax (row stats across 16-lane column groups) ----
    float alpha[8];
    #pragma unroll
    for (int r = 0; r < 8; ++r) {
      float t = fmaxf(fmaxf(sv[0][r], sv[1][r]), fmaxf(sv[2][r], sv[3][r]));
      #pragma unroll
      for (int xm = 1; xm < 16; xm <<= 1) t = fmaxf(t, __shfl_xor(t, xm, 32));
      const float mnew = fmaxf(rmax[r], t);
      alpha[r] = __expf(rmax[r] - mnew);
      rmax[r] = mnew;
    }
    #pragma unroll
    for (int r = 0; r < 8; ++r) {
      float psum = 0.f;
      #pragma unroll
      for (int sub = 0; sub < 4; ++sub) {
        const float p = __expf(sv[sub][r] - rmax[r]);
        psum += p;
        Pw[(rbase + r) * 64 + sub * 16 + m] = f2bf(p);
      }
      #pragma unroll
      for (int xm = 1; xm < 16; xm <<= 1) psum += __shfl_xor(psum, xm, 32);
      rsum[r] = rsum[r] * alpha[r] + psum;
      #pragma unroll
      for (int t2 = 0; t2 < 4; ++t2) O[t2][r] *= alpha[r];
    }
    asm volatile("s_wait_dscnt 0x0" ::: "memory");   // P visible to our reads
    // ---- reload P in A-fragment layout ----
    FragB16 ap[2];
    #pragma unroll
    for (int ks = 0; ks < 2; ++ks) {
      const uint16_t* p = Pw + m * 64 + ks * 32 + half * 8;
      ap[ks].q[0] = *(const uint4*)p;
      ap[ks].q[1] = *(const uint4*)(p + 16);
    }
    // ---- O += P . V, V-frags from LDS ----
    #pragma unroll
    for (int sub = 0; sub < 4; ++sub)
      #pragma unroll
      for (int ks = 0; ks < 2; ++ks) {
        FragB16 fv;
        const uint16_t* p = &Vs[bf][(sub * 16 + m) * 64 + ks * 32 + half * 8];
        fv.q[0] = *(const uint4*)p;
        fv.q[1] = *(const uint4*)(p + 16);
        O[sub] = __builtin_amdgcn_wmma_f32_16x16x32_bf16(
            false, ap[ks].v, false, fv.v, (short)0, O[sub], false, false);
      }

    asm volatile("s_wait_asynccnt 0x0" ::: "memory");
#if HAVE_TDM
    __builtin_amdgcn_s_wait_tensorcnt(0);
#endif
    __syncthreads();
    bf ^= 1;
  }
  // ---- normalize + store to (B,S,D) bf16 ----
  #pragma unroll
  for (int sub = 0; sub < 4; ++sub)
    #pragma unroll
    for (int r = 0; r < 8; ++r) {
      const float o = O[sub][r] / rsum[r];
      const int srow = q0 + rbase + r;
      aob[((size_t)b * S_LEN + srow) * DMODEL + h * HDIM + sub * 16 + m] = f2bf(o);
    }
}

// ---------------------------------------------------------------------------
// Kernel 4: output projection + bias -> fp32 out
// ---------------------------------------------------------------------------
__global__ __launch_bounds__(256)
void proj_gemm_kernel(const uint16_t* __restrict__ aob, const uint16_t* __restrict__ wprojb,
                      const float* __restrict__ bias, float* __restrict__ out) {
  v8f acc[2][4] = {};
  const int rowA0 = blockIdx.x * BM;
  const int rowB0 = blockIdx.y * BN;
  gemm128x128(aob, wprojb, DMODEL, rowA0, rowB0, acc);

  const int lane  = threadIdx.x & 31;
  const int wave  = threadIdx.x >> 5;
  const int wm    = (wave >> 1) * 32;
  const int wn    = (wave & 1) * 64;
  const int col   = lane & 15;
  const int rbase = (lane >> 4) * 8;
  #pragma unroll
  for (int tm = 0; tm < 2; ++tm)
    #pragma unroll
    for (int tn = 0; tn < 4; ++tn)
      #pragma unroll
      for (int r = 0; r < 8; ++r) {
        const int gm = rowA0 + wm + tm * 16 + rbase + r;
        const int gn = rowB0 + wn + tn * 16 + col;
        out[(size_t)gm * DMODEL + gn] = acc[tm][tn][r] + bias[gn];
      }
}

// ---------------------------------------------------------------------------
extern "C" void kernel_launch(void* const* d_in, const int* in_sizes, int n_in,
                              void* d_out, int out_size, void* d_ws, size_t ws_size,
                              hipStream_t stream) {
  const float* x            = (const float*)d_in[0];
  const unsigned char* mask = (const unsigned char*)d_in[1];
  const float* Wqkv         = (const float*)d_in[2];
  const float* Wproj        = (const float*)d_in[3];
  const float* bproj        = (const float*)d_in[4];
  float* out                = (float*)d_out;

  char* ws = (char*)d_ws;
  const size_t nX   = (size_t)BATCH * S_LEN * DMODEL;      // 4,194,304
  const size_t nQKV = (size_t)3 * DMODEL * DMODEL;         // 3,145,728
  const size_t nPRJ = (size_t)DMODEL * DMODEL;             // 1,048,576
  uint16_t* xb     = (uint16_t*)ws;  ws += nX   * 2;
  uint16_t* wqkvb  = (uint16_t*)ws;  ws += nQKV * 2;
  uint16_t* wprojb = (uint16_t*)ws;  ws += nPRJ * 2;
  uint16_t* qb     = (uint16_t*)ws;  ws += nX   * 2;
  uint16_t* kb     = (uint16_t*)ws;  ws += nX   * 2;
  uint16_t* vtb    = (uint16_t*)ws;  ws += nX   * 2;
  uint16_t* aob    = (uint16_t*)ws;  ws += nX   * 2;       // total ~50.3 MB

  f32_to_bf16_kernel<<<(int)(nX   / 4 / 256), 256, 0, stream>>>(x,     xb,     (int)(nX   / 4));
  f32_to_bf16_kernel<<<(int)(nQKV / 4 / 256), 256, 0, stream>>>(Wqkv,  wqkvb,  (int)(nQKV / 4));
  f32_to_bf16_kernel<<<(int)(nPRJ / 4 / 256), 256, 0, stream>>>(Wproj, wprojb, (int)(nPRJ / 4));

  qkv_gemm_kernel<<<dim3(4096 / BM, 3072 / BN), 256, 0, stream>>>(xb, wqkvb, qb, kb, vtb);

  attention_kernel<<<dim3(S_LEN / 128, BATCH * NHEAD), 256, 0, stream>>>(qb, kb, vtb, mask, aob);

  proj_gemm_kernel<<<dim3(4096 / BM, DMODEL / BN), 256, 0, stream>>>(aob, wprojb, bproj, out);
}
